// PytorchFalconDecoderModel_17514876633996
// MI455X (gfx1250) — matrix operand order repaired and checked by reading, not compiled, not run on hardware
//
#include <hip/hip_runtime.h>
#include <hip/hip_bf16.h>

// ---------------------------------------------------------------------------
// Falcon decoder layer for MI455X (gfx1250, wave32, WMMA 16x16x32 bf16).
// GEMMs are HBM-bound on fp32 weight streaming (~830MB -> ~36us @23.3TB/s):
//  - grid is M-major so the 8 blocks sharing a 64-wide weight strip are
//    dispatch-adjacent and hit the 192MB L2 (weight traffic ~1x, not 8x)
//  - weights are converted fp32->bf16 in registers while staging to LDS
//  - bf16 activation tiles use GLOBAL_LOAD_ASYNC_TO_LDS_B128 (ASYNCcnt) with
//    double-buffered LDS; weight stream gets global_prefetch hints
// Attention is flash-style online softmax with WMMA for qK^T and pV.
// ---------------------------------------------------------------------------

typedef __bf16 bf16_t;
typedef __attribute__((ext_vector_type(16))) __bf16 v16bf;
typedef __attribute__((ext_vector_type(8)))  float  v8f;
typedef __attribute__((ext_vector_type(4)))  unsigned int uint4v;
typedef __attribute__((__vector_size__(4 * sizeof(int)))) int v4i;

#define S_LEN   1024
#define H_DIM   4544
#define NHEADS  71
#define HDIM    64
#define FFN_DIM 18176
#define NQKV    (H_DIM + 2 * HDIM)   // 4672

#if defined(__has_builtin)
#  if __has_builtin(__builtin_amdgcn_global_load_async_to_lds_b128)
#    define HAVE_ASYNC_COPY 1
#  endif
#endif
#ifndef HAVE_ASYNC_COPY
#  define HAVE_ASYNC_COPY 0
#endif

// Per-lane 16B global->LDS copy. Async path is tracked by ASYNCcnt.
// Builtin signature (from toolchain diagnostic): (v4i AS1*, v4i AS3*, Ii, Ii).
__device__ __forceinline__ void copy_b128(const bf16_t* g, bf16_t* l) {
#if HAVE_ASYNC_COPY
  __builtin_amdgcn_global_load_async_to_lds_b128(
      (__attribute__((address_space(1))) v4i*)g,
      (__attribute__((address_space(3))) v4i*)l, 0, 0);
#else
  *(uint4v*)l = *(const uint4v*)g;
#endif
}
__device__ __forceinline__ void wait_async_all() {
#if HAVE_ASYNC_COPY
  asm volatile("s_wait_asynccnt 0x0" ::: "memory");
#endif
}

__device__ __forceinline__ v8f wmma_bf16(v16bf a, v16bf b, v8f c) {
  return __builtin_amdgcn_wmma_f32_16x16x32_bf16(
      /*neg_a=*/false, a, /*neg_b=*/false, b,
      /*c_mod=*/(short)0, c, /*reuse_a=*/false, /*reuse_b=*/false);
}

// A fragment (16x32 bf16, ISA 7.12.2): lane = h*16 + r holds row M=r,
// element e <-> K = e + (e<8 ? 0 : 8) + 8*h  ->  two 16B contiguous chunks.
__device__ __forceinline__ v16bf load_a(const bf16_t* rowp, int h) {
  union { uint4v u[2]; v16bf v; } t;
  t.u[0] = *(const uint4v*)(rowp + 8 * h);
  t.u[1] = *(const uint4v*)(rowp + 8 * h + 16);
  return t.v;
}

// B fragment (32x16 bf16): lane = h*16 + c holds col N=c,
// element e <-> K = e + (e<8 ? 0 : 8) + 8*h. LDS keeps B transposed ([n][k]).
__device__ __forceinline__ v16bf load_b(const bf16_t* colp, int h) {
  union { uint4v u[2]; v16bf v; } t;
  t.u[0] = *(const uint4v*)(colp + 8 * h);
  t.u[1] = *(const uint4v*)(colp + 8 * h + 16);
  return t.v;
}

// ------------------------------- LayerNorm ---------------------------------
__global__ __launch_bounds__(256) void ln_kernel(
    const float* __restrict__ x, const float* __restrict__ w,
    const float* __restrict__ b, bf16_t* __restrict__ out) {
  __shared__ float xs[H_DIM];
  __shared__ float red[16];
  const int s = blockIdx.x, tid = threadIdx.x;
  const float* xr = x + (size_t)s * H_DIM;
  float sum = 0.f, sq = 0.f;
  for (int i = tid; i < H_DIM; i += 256) {
    float v = xr[i]; xs[i] = v; sum += v; sq += v * v;
  }
  for (int off = 16; off >= 1; off >>= 1) {
    sum += __shfl_xor(sum, off, 32);
    sq  += __shfl_xor(sq,  off, 32);
  }
  const int wid = tid >> 5, lane = tid & 31;
  if (lane == 0) { red[wid] = sum; red[8 + wid] = sq; }
  __syncthreads();
  sum = 0.f; sq = 0.f;
#pragma unroll
  for (int i = 0; i < 8; i++) { sum += red[i]; sq += red[8 + i]; }
  const float mean = sum * (1.0f / H_DIM);
  const float var  = sq * (1.0f / H_DIM) - mean * mean;
  const float rstd = rsqrtf(var + 1e-5f);
  for (int i = tid; i < H_DIM; i += 256) {
    float v = (xs[i] - mean) * rstd * w[i] + b[i];
    out[(size_t)s * H_DIM + i] = (bf16_t)v;
  }
}

// ---------------- Generic WMMA GEMM, 128x64 tile, double-buffered ----------
// C[M,N] = epilogue(A_bf16[M,K] @ W_f32[K,N])
// MODE 0: store bf16   MODE 1: exact GELU -> bf16
// MODE 2: store f32    MODE 3: out = R0 + R1 + acc (f32)
template <int MODE>
__global__ __launch_bounds__(256) void gemm_kernel(
    const bf16_t* __restrict__ A, const float* __restrict__ W,
    int M, int N, int K,
    bf16_t* __restrict__ Obf, float* __restrict__ Of,
    const float* __restrict__ R0, const float* __restrict__ R1) {
  (void)M;
  __shared__ __align__(16) bf16_t As[2][128 * 40];  // [m][k], pad 40 halves
  __shared__ __align__(16) bf16_t Bs[2][64 * 40];   // [n][k] transposed
  const int tid = threadIdx.x;
  const int m0 = blockIdx.x * 128;      // M on x: strip-sharing blocks adjacent
  const int n0 = blockIdx.y * 64;
  const int wid = tid >> 5, lane = tid & 31;
  const int wm = wid & 3, wn = wid >> 2;            // 4x2 waves -> 32x32 each
  const int h = lane >> 4, c = lane & 15;
  v8f acc[2][2] = {};
  const int NT = K >> 5;

  // A tile copy plan: 128 rows x 64B = 512 x 16B chunks, 2 per thread.
  const int ar0 = tid >> 1, aj0 = (tid & 1) * 2;    // rows ar0, chunks aj0..aj0+1
  // B tile plan: 2048 f32, 8 per thread: i = t*256+tid -> k=i>>6, n=i&63.

  // ---- prologue: stage tile 0 into buffer 0 ----
  {
#pragma unroll
    for (int j = 0; j < 2; j++)
      copy_b128(A + (size_t)(m0 + ar0) * K + (aj0 + j) * 8,
                &As[0][ar0 * 40 + (aj0 + j) * 8]);
#pragma unroll
    for (int t = 0; t < 8; t++) {
      int i = t * 256 + tid, kk = i >> 6, nn = i & 63;
      Bs[0][nn * 40 + kk] = (bf16_t)W[(size_t)kk * N + n0 + nn];
    }
  }
  wait_async_all();
  __syncthreads();

  for (int kt = 0; kt < NT; kt++) {
    const int cur = kt & 1, nxt = cur ^ 1;
    const bool pf = (kt + 1 < NT);
    float breg[8];
    if (pf) {
      const int k1 = (kt + 1) * 32;
#pragma unroll
      for (int j = 0; j < 2; j++)                   // async A (ASYNCcnt)
        copy_b128(A + (size_t)(m0 + ar0) * K + k1 + (aj0 + j) * 8,
                  &As[nxt][ar0 * 40 + (aj0 + j) * 8]);
#pragma unroll
      for (int t = 0; t < 8; t++) {                 // B fp32 -> regs
        int i = t * 256 + tid, kk = i >> 6, nn = i & 63;
        breg[t] = W[(size_t)(k1 + kk) * N + n0 + nn];
      }
      if (kt + 2 < NT)                              // weight stream prefetch
        __builtin_prefetch(W + (size_t)((kt + 2) * 32 + (tid >> 6)) * N +
                               n0 + (tid & 63), 0, 1);
    }

    // compute on current buffer
    v16bf a0 = load_a(&As[cur][(wm * 32 + c) * 40], h);
    v16bf a1 = load_a(&As[cur][(wm * 32 + 16 + c) * 40], h);
    v16bf b0 = load_b(&Bs[cur][(wn * 32 + c) * 40], h);
    v16bf b1 = load_b(&Bs[cur][(wn * 32 + 16 + c) * 40], h);
    acc[0][0] = wmma_bf16(a0, b0, acc[0][0]);
    acc[0][1] = wmma_bf16(a0, b1, acc[0][1]);
    acc[1][0] = wmma_bf16(a1, b0, acc[1][0]);
    acc[1][1] = wmma_bf16(a1, b1, acc[1][1]);

    if (pf) {
#pragma unroll
      for (int t = 0; t < 8; t++) {                 // cvt + stage B
        int i = t * 256 + tid, kk = i >> 6, nn = i & 63;
        Bs[nxt][nn * 40 + kk] = (bf16_t)breg[t];
      }
    }
    wait_async_all();
    __syncthreads();
  }

#pragma unroll
  for (int i = 0; i < 2; i++) {
#pragma unroll
    for (int j = 0; j < 2; j++) {
      const int rbase = m0 + wm * 32 + i * 16 + h * 8;
      const int col = n0 + wn * 32 + j * 16 + c;
#pragma unroll
      for (int v = 0; v < 8; v++) {
        const size_t idx = (size_t)(rbase + v) * N + col;
        float val = acc[i][j][v];
        if (MODE == 0) {
          Obf[idx] = (bf16_t)val;
        } else if (MODE == 1) {
          float g = 0.5f * val * (1.0f + erff(val * 0.70710678f));
          Obf[idx] = (bf16_t)g;
        } else if (MODE == 2) {
          Of[idx] = val;
        } else {
          Of[idx] = R0[idx] + R1[idx] + val;
        }
      }
    }
  }
}

// --------------------- Flash attention (multi-query) -----------------------
// Per block: 1 head x 128 query rows; 8 waves, 16 q-rows/wave; stream K/V in
// 32-key tiles up to the causal bound with online softmax.
__global__ __launch_bounds__(256) void attn_kernel(
    const bf16_t* __restrict__ qkv, const float* __restrict__ alibi,
    bf16_t* __restrict__ ctx) {
  __shared__ __align__(16) bf16_t Qs[128 * 72];      // [qrow][d], pad 72
  __shared__ __align__(16) bf16_t Ks[32 * 72];       // [kpos][d]
  __shared__ __align__(16) bf16_t Vs[64 * 40];       // [d][kpos] (transposed)
  __shared__ __align__(16) bf16_t Ps[8 * 16 * 40];   // per-wave p scratch
  const int head = blockIdx.x;
  const int q0 = blockIdx.y * 128;
  const int tid = threadIdx.x, wid = tid >> 5, lane = tid & 31;
  const int h = lane >> 4, c = lane & 15;

  // Q tile: 128 rows x 128B = 1024 x 16B chunks, 4 per thread (async).
  for (int i = tid; i < 1024; i += 256) {
    int r = i >> 3, j = i & 7;
    copy_b128(qkv + (size_t)(q0 + r) * NQKV + head * HDIM + j * 8,
              &Qs[r * 72 + j * 8]);
  }
  wait_async_all();
  __syncthreads();
  v16bf aq0 = load_a(&Qs[(wid * 16 + c) * 72], h);        // d = 0..31
  v16bf aq1 = load_a(&Qs[(wid * 16 + c) * 72 + 32], h);   // d = 32..63

  v8f acc[4] = {};
  float mrow[8], lrow[8];
#pragma unroll
  for (int v = 0; v < 8; v++) { mrow[v] = -3.0e38f; lrow[v] = 0.f; }
  const float* al_h = alibi + (size_t)head * S_LEN;
  bf16_t* Pw = &Ps[wid * 16 * 40];
  const int nkb = (q0 + 128) >> 5;                   // causal bound

  for (int kb = 0; kb < nkb; kb++) {
    __syncthreads();
    {                                                // K tile: 256 chunks async
      int r = tid >> 3, j = tid & 7;
      copy_b128(qkv + (size_t)(kb * 32 + r) * NQKV + H_DIM + j * 8,
                &Ks[r * 72 + j * 8]);
    }
    for (int i = tid; i < 32 * 64; i += 256) {       // V tile transposed [d][kpos]
      int kp = i & 31, d = i >> 5;
      Vs[d * 40 + kp] = qkv[(size_t)(kb * 32 + kp) * NQKV + H_DIM + HDIM + d];
    }
    wait_async_all();
    __syncthreads();

    float logit[2][8], tmax[8];
#pragma unroll
    for (int v = 0; v < 8; v++) tmax[v] = -3.0e38f;
#pragma unroll
    for (int nb = 0; nb < 2; nb++) {                 // s = q @ k^T (K=64)
      v16bf bk0 = load_b(&Ks[(nb * 16 + c) * 72], h);
      v16bf bk1 = load_b(&Ks[(nb * 16 + c) * 72 + 32], h);
      v8f t = {};
      t = wmma_bf16(aq0, bk0, t);
      t = wmma_bf16(aq1, bk1, t);
      const int col = kb * 32 + nb * 16 + c;
      const float al = al_h[col];
#pragma unroll
      for (int v = 0; v < 8; v++) {
        int row = q0 + wid * 16 + h * 8 + v;
        float lg = (t[v] + al) * 0.125f + (col <= row ? 0.f : -1.0e9f);
        logit[nb][v] = lg;
        tmax[v] = fmaxf(tmax[v], lg);
      }
    }
#pragma unroll
    for (int v = 0; v < 8; v++) {                    // online softmax per row
      float m = tmax[v];
      for (int off = 1; off < 16; off <<= 1) m = fmaxf(m, __shfl_xor(m, off, 32));
      float mn = fmaxf(mrow[v], m);
      float scale = __expf(mrow[v] - mn);
      mrow[v] = mn;
      float ps = 0.f;
#pragma unroll
      for (int nb = 0; nb < 2; nb++) {
        float p = __expf(logit[nb][v] - mn);
        logit[nb][v] = p;
        ps += p;
      }
      for (int off = 1; off < 16; off <<= 1) ps += __shfl_xor(ps, off, 32);
      lrow[v] = lrow[v] * scale + ps;
#pragma unroll
      for (int j = 0; j < 4; j++) acc[j][v] *= scale;
#pragma unroll
      for (int nb = 0; nb < 2; nb++)                 // p -> LDS (A layout src)
        Pw[(h * 8 + v) * 40 + nb * 16 + c] = (bf16_t)logit[nb][v];
    }
    // In-wave LDS RAW: compiler inserts s_wait_dscnt; no barrier needed.
    v16bf ap = load_a(&Pw[c * 40], h);
#pragma unroll
    for (int j = 0; j < 4; j++) {                    // ctx += p @ v
      v16bf bv = load_b(&Vs[(j * 16 + c) * 40], h);
      acc[j] = wmma_bf16(ap, bv, acc[j]);
    }
  }

#pragma unroll
  for (int j = 0; j < 4; j++)
#pragma unroll
    for (int v = 0; v < 8; v++) {
      int row = q0 + wid * 16 + h * 8 + v;
      ctx[(size_t)row * H_DIM + head * HDIM + j * 16 + c] =
          (bf16_t)(acc[j][v] * (1.0f / lrow[v]));
    }
}

// ------------------------------- launcher ----------------------------------
extern "C" void kernel_launch(void* const* d_in, const int* in_sizes, int n_in,
                              void* d_out, int out_size, void* d_ws,
                              size_t ws_size, hipStream_t stream) {
  (void)in_sizes; (void)n_in; (void)out_size; (void)ws_size;
  const float* x       = (const float*)d_in[0];
  const float* alibi   = (const float*)d_in[1];
  // d_in[2] attention_mask: causal + (-1e9) applied analytically in-kernel.
  const float* ln_w    = (const float*)d_in[3];
  const float* ln_b    = (const float*)d_in[4];
  const float* w_qkv   = (const float*)d_in[5];
  const float* w_dense = (const float*)d_in[6];
  const float* w_fc    = (const float*)d_in[7];
  const float* w_proj  = (const float*)d_in[8];
  float* out = (float*)d_out;

  char* ws = (char*)d_ws;                            // ~84 MB total
  bf16_t* ln_bf  = (bf16_t*)ws; ws += (size_t)S_LEN * H_DIM * 2;
  bf16_t* qkv_bf = (bf16_t*)ws; ws += (size_t)S_LEN * NQKV * 2;
  bf16_t* ctx_bf = (bf16_t*)ws; ws += (size_t)S_LEN * H_DIM * 2;
  bf16_t* h_bf   = (bf16_t*)ws; ws += (size_t)S_LEN * FFN_DIM * 2;
  float*  attn_f = (float*)ws;  ws += (size_t)S_LEN * H_DIM * 4;

  ln_kernel<<<S_LEN, 256, 0, stream>>>(x, ln_w, ln_b, ln_bf);
  gemm_kernel<0><<<dim3(S_LEN / 128, NQKV / 64), 256, 0, stream>>>(
      ln_bf, w_qkv, S_LEN, NQKV, H_DIM, qkv_bf, nullptr, nullptr, nullptr);
  attn_kernel<<<dim3(NHEADS, S_LEN / 128), 256, 0, stream>>>(
      qkv_bf, alibi, ctx_bf);
  gemm_kernel<2><<<dim3(S_LEN / 128, H_DIM / 64), 256, 0, stream>>>(
      ctx_bf, w_dense, S_LEN, H_DIM, H_DIM, nullptr, attn_f, nullptr, nullptr);
  gemm_kernel<1><<<dim3(S_LEN / 128, FFN_DIM / 64), 256, 0, stream>>>(
      ln_bf, w_fc, S_LEN, FFN_DIM, H_DIM, h_bf, nullptr, nullptr, nullptr);
  gemm_kernel<3><<<dim3(S_LEN / 128, H_DIM / 64), 256, 0, stream>>>(
      h_bf, w_proj, S_LEN, H_DIM, FFN_DIM, nullptr, out, x, attn_f);
}
